// SoftShapeLayer_36507222016566
// MI455X (gfx1250) — matrix operand order repaired and checked by reading, not compiled, not run on hardware
//
#include <hip/hip_runtime.h>
#include <hip/hip_bf16.h>
#include <math.h>

typedef __attribute__((ext_vector_type(16))) _Float16 v16h;
typedef __attribute__((ext_vector_type(8)))  float    v8f;

#define BB 4
#define NN 4096
#define DD 512
#define EE 4
#define HH 1024
#define NFNF 32
#define KLEFT 2048
#define TT 2049
#define RR 8196      // BB*TT

#define BLK_M 128
#define BLK_N 64
#define BLK_K 32
#define A_ST 40      // LDS row stride (halves) for A tile rows (16B aligned)
#define B_ST 40      // LDS column stride (halves) for transposed B tile

// ---------------- CDNA5 async global->LDS copy (ASYNCcnt path) ----------------
#if __has_builtin(__builtin_amdgcn_global_load_async_to_lds_b128)
#define HAVE_ASYNC_LDS 1
typedef int v4i_b128 __attribute__((vector_size(16)));
typedef __attribute__((address_space(1))) v4i_b128* gv4i_p;   // global (AS1)
typedef __attribute__((address_space(3))) v4i_b128* lv4i_p;   // LDS (AS3)
#endif

__device__ __forceinline__ void async_copy16(const _Float16* g, _Float16* l) {
#ifdef HAVE_ASYNC_LDS
  __builtin_amdgcn_global_load_async_to_lds_b128(
      (gv4i_p)(uintptr_t)g,
      (lv4i_p)(unsigned int)(uintptr_t)l,
      0, 0);
#else
  *reinterpret_cast<float4*>(l) = *reinterpret_cast<const float4*>(g);
#endif
}

__device__ __forceinline__ void async_wait0() {
#ifdef HAVE_ASYNC_LDS
#if __has_builtin(__builtin_amdgcn_s_wait_asynccnt)
  __builtin_amdgcn_s_wait_asynccnt(0);
#else
  asm volatile("s_wait_asynccnt 0x0" ::: "memory");
#endif
#endif
}

__device__ __forceinline__ float gelu_f(float x) {
  return 0.5f * x * (1.0f + erff(x * 0.7071067811865475f));
}

__device__ __forceinline__ v8f vz8() {
  v8f z;
#pragma unroll
  for (int i = 0; i < 8; ++i) z[i] = 0.0f;
  return z;
}

__device__ __forceinline__ float blk_reduce256(float v, float* sh, int tid) {
  sh[tid] = v;
  __syncthreads();
#pragma unroll
  for (int s = 128; s > 0; s >>= 1) {
    if (tid < s) sh[tid] += sh[tid + s];
    __syncthreads();
  }
  float r = sh[0];
  __syncthreads();
  return r;
}

// ================= shared WMMA GEMM core =================
// 8 waves (4 in M, 2 in N); wave tile 32x32 = 4 x v_wmma_f32_16x16x32_f16.
// A: row-major (M,K) f16; B: row-major (K,N) f16; K mult of 32, N mult of 64.
// A tile staged row-major via async b128 loads; B tile staged TRANSPOSED
// (Bsh[n][k]) so every fragment read is 2 contiguous ds_load_b128.
__device__ __forceinline__ void gemm_core(
    const _Float16* __restrict__ A, const _Float16* __restrict__ B,
    int M, int K, int N, int row0, int col0, int tid,
    _Float16* Ash, _Float16* Bsh, v8f acc[2][2]) {
  const int lane = tid & 31, wave = tid >> 5;
  const int wm = wave & 3, wn = wave >> 2;
  const int half = lane >> 4, lr = lane & 15;
  const bool fullM = (row0 + BLK_M <= M);

  for (int k0 = 0; k0 < K; k0 += BLK_K) {
    // ---- stage A tile (BLK_M x BLK_K): 512 chunks of 16B, 2 per thread ----
    if (fullM) {
      for (int c = tid; c < BLK_M * 4; c += 256) {
        int r = c >> 2, cc = c & 3;
        async_copy16(A + (size_t)(row0 + r) * K + k0 + cc * 8,
                     &Ash[r * A_ST + cc * 8]);
      }
      if (k0 + BLK_K < K)
        __builtin_prefetch(A + (size_t)(row0 + (tid >> 1)) * K + k0 + BLK_K, 0, 1);
    } else {
      for (int c = tid; c < BLK_M * 4; c += 256) {
        int r = c >> 2, cc = c & 3;
        float4 v = make_float4(0.f, 0.f, 0.f, 0.f);
        if (row0 + r < M)
          v = *reinterpret_cast<const float4*>(A + (size_t)(row0 + r) * K + k0 + cc * 8);
        *reinterpret_cast<float4*>(&Ash[r * A_ST + cc * 8]) = v;
      }
    }
    // ---- stage B tile transposed: thread (kp,cc) packs rows 2kp,2kp+1 ----
    if (tid < 128) {
      int kp = tid >> 3, cc = tid & 7;
      const _Float16* src0 = B + (size_t)(k0 + 2 * kp) * N + col0 + cc * 8;
      float4 r0 = *reinterpret_cast<const float4*>(src0);
      float4 r1 = *reinterpret_cast<const float4*>(src0 + N);
      const unsigned short* h0 = reinterpret_cast<const unsigned short*>(&r0);
      const unsigned short* h1 = reinterpret_cast<const unsigned short*>(&r1);
#pragma unroll
      for (int j = 0; j < 8; ++j) {
        int n = cc * 8 + j;
        unsigned pack = (unsigned)h0[j] | ((unsigned)h1[j] << 16);
        *reinterpret_cast<unsigned*>(&Bsh[n * B_ST + 2 * kp]) = pack;
      }
    }
    async_wait0();
    __syncthreads();

    // ---- fragments: 2 x ds_load_b128 each ----
    union AB { v16h v; float4 f[2]; };
    AB af[2], bf[2];
#pragma unroll
    for (int mi = 0; mi < 2; ++mi) {
      int r = wm * 32 + mi * 16 + lr;
      af[mi].f[0] = *reinterpret_cast<const float4*>(&Ash[r * A_ST + half * 8]);
      af[mi].f[1] = *reinterpret_cast<const float4*>(&Ash[r * A_ST + 16 + half * 8]);
    }
#pragma unroll
    for (int ni = 0; ni < 2; ++ni) {
      int n = wn * 32 + ni * 16 + lr;
      bf[ni].f[0] = *reinterpret_cast<const float4*>(&Bsh[n * B_ST + half * 8]);
      bf[ni].f[1] = *reinterpret_cast<const float4*>(&Bsh[n * B_ST + 16 + half * 8]);
    }
#pragma unroll
    for (int mi = 0; mi < 2; ++mi)
#pragma unroll
      for (int ni = 0; ni < 2; ++ni)
        acc[mi][ni] = __builtin_amdgcn_wmma_f32_16x16x32_f16(
            false, af[mi].v, false, bf[ni].v, (short)0, acc[mi][ni], false, false);
    __syncthreads();
  }
}

// ---------------- RMSNorm + sigmoid score + xs ----------------
__global__ __launch_bounds__(256) void k_rms_score(
    const float* __restrict__ x, const float* __restrict__ gamma,
    const float* __restrict__ attn_w, const float* __restrict__ attn_b,
    float* __restrict__ xs, float* __restrict__ scores) {
  __shared__ float sh[256];
  __shared__ float s_bcast;
  int row = blockIdx.x;
  int tid = threadIdx.x;
  const float* xr = x + (size_t)row * DD;
  float ss = 0.0f;
  for (int d = tid; d < DD; d += 256) { float v = xr[d]; ss += v * v; }
  float sum = blk_reduce256(ss, sh, tid);
  float inv = 22.627416997969522f / fmaxf(sqrtf(sum), 1e-12f); // sqrt(512)/||x||
  float dp = 0.0f;
  for (int d = tid; d < DD; d += 256) dp += xr[d] * inv * gamma[d] * attn_w[d];
  float dot = blk_reduce256(dp, sh, tid);
  if (tid == 0) {
    float sc = 1.0f / (1.0f + expf(-(dot + attn_b[0])));
    scores[row] = sc;
    s_bcast = sc;
  }
  __syncthreads();
  float sc = s_bcast;
  for (int d = tid; d < DD; d += 256)
    xs[(size_t)row * DD + d] = xr[d] * inv * gamma[d] * sc;
}

// ---------------- exact top-k via rank counting (ties -> lower index) -------
__global__ __launch_bounds__(256) void k_rank(const float* __restrict__ scores,
                                              int* __restrict__ mask) {
  __shared__ int sh[256];
  int idx = blockIdx.x;
  int b = idx >> 12, n = idx & (NN - 1);
  int tid = threadIdx.x;
  const float* s = scores + (size_t)b * NN;
  float si = s[n];
  int cnt = 0;
  for (int j = tid; j < NN; j += 256) {
    float sj = s[j];
    cnt += (sj > si) || (sj == si && j < n);
  }
  sh[tid] = cnt;
  __syncthreads();
  for (int st = 128; st > 0; st >>= 1) {
    if (tid < st) sh[tid] += sh[tid + st];
    __syncthreads();
  }
  if (tid == 0) mask[idx] = (sh[0] < KLEFT) ? 1 : 0;
}

// ---------------- compact selected indices (index-sorted order) -------------
__global__ __launch_bounds__(1024) void k_compact(const int* __restrict__ mask,
                                                  int* __restrict__ idxmap) {
  __shared__ int sh[1024];
  int b = blockIdx.x, tid = threadIdx.x;
  int base = b * NN + tid * 4;
  int m0 = mask[base], m1 = mask[base + 1], m2 = mask[base + 2], m3 = mask[base + 3];
  int lsum = m0 + m1 + m2 + m3;
  sh[tid] = lsum;
  __syncthreads();
  for (int off = 1; off < 1024; off <<= 1) {
    int add = (tid >= off) ? sh[tid - off] : 0;
    __syncthreads();
    sh[tid] += add;
    __syncthreads();
  }
  int pos = sh[tid] - lsum;  // exclusive prefix
  int n = tid * 4;
  if (m0) idxmap[b * KLEFT + pos++] = n;
  if (m1) idxmap[b * KLEFT + pos++] = n + 1;
  if (m2) idxmap[b * KLEFT + pos++] = n + 2;
  if (m3) idxmap[b * KLEFT + pos++] = n + 3;
}

// ---------------- complement (non-selected) row sum ----------------
__global__ __launch_bounds__(512) void k_extra(const float* __restrict__ xs,
                                               const int* __restrict__ mask,
                                               float* __restrict__ extra) {
  int b = blockIdx.x, d = threadIdx.x;
  float acc = 0.0f;
  for (int n = 0; n < NN; ++n)
    if (!mask[b * NN + n]) acc += xs[(((size_t)b * NN) + n) * DD + d];
  extra[b * DD + d] = acc;
}

// ---------------- build xcat = [gathered left_x ; extra] ----------------
__global__ void k_xcat(const float* __restrict__ xs, const int* __restrict__ idxmap,
                       const float* __restrict__ extra, float* __restrict__ xcat) {
  int idx = blockIdx.x * 256 + threadIdx.x;
  if (idx >= RR * DD) return;
  int d = idx & (DD - 1);
  int row = idx / DD;
  int b = row / TT, t = row % TT;
  float v;
  if (t < KLEFT) {
    int n = idxmap[b * KLEFT + t];
    v = xs[(((size_t)b * NN) + n) * DD + d];
  } else {
    v = extra[b * DD + d];
  }
  xcat[idx] = v;
}

// ---------------- second RMSNorm ----------------
__global__ __launch_bounds__(256) void k_rms2(const float* __restrict__ xcat,
                                              const float* __restrict__ gamma,
                                              float* __restrict__ xn2) {
  __shared__ float sh[256];
  int row = blockIdx.x, tid = threadIdx.x;
  const float* xr = xcat + (size_t)row * DD;
  float ss = 0.0f;
  for (int d = tid; d < DD; d += 256) { float v = xr[d]; ss += v * v; }
  float sum = blk_reduce256(ss, sh, tid);
  float inv = 22.627416997969522f / fmaxf(sqrtf(sum), 1e-12f);
  for (int d = tid; d < DD; d += 256)
    xn2[(size_t)row * DD + d] = xr[d] * inv * gamma[d];
}

// ---------------- elementwise helpers ----------------
__global__ void k_cvt(const float* __restrict__ s, _Float16* __restrict__ d, int n) {
  int i = blockIdx.x * 256 + threadIdx.x;
  if (i < n) d[i] = (_Float16)s[i];
}

__global__ void k_zero(float* __restrict__ p, int n) {
  int i = blockIdx.x * 256 + threadIdx.x;
  if (i < n) p[i] = 0.0f;
}

// ---------------- gate softmax + mean accumulation ----------------
__global__ __launch_bounds__(256) void k_gate(const float* __restrict__ xn2,
                                              const float* __restrict__ gate_w,
                                              float* __restrict__ gate,
                                              float* __restrict__ gate_sum) {
  __shared__ float sh[256];
  __shared__ float logit[EE];
  int row = blockIdx.x, tid = threadIdx.x;
  const float* xr = xn2 + (size_t)row * DD;
  float l[EE] = {0.f, 0.f, 0.f, 0.f};
  for (int d = tid; d < DD; d += 256) {
    float v = xr[d];
#pragma unroll
    for (int e = 0; e < EE; ++e) l[e] += v * gate_w[d * EE + e];
  }
  for (int e = 0; e < EE; ++e) {
    float t = blk_reduce256(l[e], sh, tid);
    if (tid == 0) logit[e] = t;
  }
  __syncthreads();
  if (tid == 0) {
    float m = fmaxf(fmaxf(logit[0], logit[1]), fmaxf(logit[2], logit[3]));
    float p[EE], s = 0.0f;
    for (int e = 0; e < EE; ++e) { p[e] = expf(logit[e] - m); s += p[e]; }
    for (int e = 0; e < EE; ++e) {
      p[e] /= s;
      gate[(size_t)row * EE + e] = p[e];
      atomicAdd(&gate_sum[e], p[e]);
    }
  }
}

// GEMM1: h[e] = gelu(xn2 @ w1[e]) -> f16
__global__ __launch_bounds__(256) void k_gemm1(
    const _Float16* __restrict__ Aall, const _Float16* __restrict__ W1,
    _Float16* __restrict__ Hb, int M, int K, int N) {
  __shared__ __align__(16) _Float16 Ash[BLK_M * A_ST];
  __shared__ __align__(16) _Float16 Bsh[BLK_N * B_ST];
  const int tid = threadIdx.x;
  const int lane = tid & 31, wave = tid >> 5;
  const int wm = wave & 3, wn = wave >> 2;
  const int half = lane >> 4, lr = lane & 15;
  const int row0 = blockIdx.y * BLK_M;
  const int col0 = blockIdx.x * BLK_N;
  const _Float16* B = W1 + (size_t)blockIdx.z * K * N;
  _Float16* O = Hb + (size_t)blockIdx.z * (size_t)M * N;

  v8f acc[2][2];
#pragma unroll
  for (int i = 0; i < 2; ++i)
#pragma unroll
    for (int j = 0; j < 2; ++j) acc[i][j] = vz8();

  gemm_core(Aall, B, M, K, N, row0, col0, tid, Ash, Bsh, acc);

#pragma unroll
  for (int mi = 0; mi < 2; ++mi)
#pragma unroll
    for (int ni = 0; ni < 2; ++ni) {
      int col = col0 + wn * 32 + ni * 16 + lr;
#pragma unroll
      for (int r = 0; r < 8; ++r) {
        int row = row0 + wm * 32 + mi * 16 + (half << 3) + r;
        if (row < M) O[(size_t)row * N + col] = (_Float16)gelu_f(acc[mi][ni][r]);
      }
    }
}

// GEMM2 fused: moe_out = sum_e gate[:,e] * (h[e] @ w2[e])
__global__ __launch_bounds__(256) void k_gemm2(
    const _Float16* __restrict__ Hb, const _Float16* __restrict__ W2,
    const float* __restrict__ gate, float* __restrict__ moe_out,
    int M, int K, int N) {
  __shared__ __align__(16) _Float16 Ash[BLK_M * A_ST];
  __shared__ __align__(16) _Float16 Bsh[BLK_N * B_ST];
  __shared__ float gsh[BLK_M][EE];
  const int tid = threadIdx.x;
  const int lane = tid & 31, wave = tid >> 5;
  const int wm = wave & 3, wn = wave >> 2;
  const int half = lane >> 4, lr = lane & 15;
  const int row0 = blockIdx.y * BLK_M;
  const int col0 = blockIdx.x * BLK_N;

  for (int c = tid; c < BLK_M * EE; c += 256) {
    int r = c >> 2, e = c & 3;
    gsh[r][e] = (row0 + r < M) ? gate[(size_t)(row0 + r) * EE + e] : 0.0f;
  }
  __syncthreads();

  v8f tot[2][2];
#pragma unroll
  for (int i = 0; i < 2; ++i)
#pragma unroll
    for (int j = 0; j < 2; ++j) tot[i][j] = vz8();

  for (int e = 0; e < EE; ++e) {
    const _Float16* A = Hb + (size_t)e * (size_t)M * K;
    const _Float16* B = W2 + (size_t)e * K * N;
    v8f acc[2][2];
#pragma unroll
    for (int i = 0; i < 2; ++i)
#pragma unroll
      for (int j = 0; j < 2; ++j) acc[i][j] = vz8();

    gemm_core(A, B, M, K, N, row0, col0, tid, Ash, Bsh, acc);

#pragma unroll
    for (int mi = 0; mi < 2; ++mi)
#pragma unroll
      for (int ni = 0; ni < 2; ++ni)
#pragma unroll
        for (int r = 0; r < 8; ++r) {
          int rl = wm * 32 + mi * 16 + (half << 3) + r;
          tot[mi][ni][r] += acc[mi][ni][r] * gsh[rl][e];
        }
  }

#pragma unroll
  for (int mi = 0; mi < 2; ++mi)
#pragma unroll
    for (int ni = 0; ni < 2; ++ni) {
      int col = col0 + wn * 32 + ni * 16 + lr;
#pragma unroll
      for (int r = 0; r < 8; ++r) {
        int row = row0 + wm * 32 + mi * 16 + (half << 3) + r;
        if (row < M) moe_out[(size_t)row * N + col] = tot[mi][ni][r];
      }
    }
}

// Projection GEMM with fused residual + gelu: out = gelu(xcat + moe + A@B + b)
__global__ __launch_bounds__(256) void k_gemmproj(
    const _Float16* __restrict__ A, const _Float16* __restrict__ B,
    const float* __restrict__ proj_b, const float* __restrict__ xcat,
    const float* __restrict__ moe_out, float* __restrict__ out,
    int M, int K, int N) {
  __shared__ __align__(16) _Float16 Ash[BLK_M * A_ST];
  __shared__ __align__(16) _Float16 Bsh[BLK_N * B_ST];
  const int tid = threadIdx.x;
  const int lane = tid & 31, wave = tid >> 5;
  const int wm = wave & 3, wn = wave >> 2;
  const int half = lane >> 4, lr = lane & 15;
  const int row0 = blockIdx.y * BLK_M;
  const int col0 = blockIdx.x * BLK_N;

  v8f acc[2][2];
#pragma unroll
  for (int i = 0; i < 2; ++i)
#pragma unroll
    for (int j = 0; j < 2; ++j) acc[i][j] = vz8();

  gemm_core(A, B, M, K, N, row0, col0, tid, Ash, Bsh, acc);

#pragma unroll
  for (int mi = 0; mi < 2; ++mi)
#pragma unroll
    for (int ni = 0; ni < 2; ++ni) {
      int col = col0 + wn * 32 + ni * 16 + lr;
#pragma unroll
      for (int r = 0; r < 8; ++r) {
        int row = row0 + wm * 32 + mi * 16 + (half << 3) + r;
        if (row < M) {
          size_t o = (size_t)row * N + col;
          float v = acc[mi][ni][r] + proj_b[col] + xcat[o] + moe_out[o];
          out[o] = gelu_f(v);
        }
      }
    }
}

// ---------------- Inception branch ----------------
__global__ void k_mpx(const float* __restrict__ xn2, float* __restrict__ mpx) {
  int idx = blockIdx.x * 256 + threadIdx.x;
  if (idx >= RR * DD) return;
  int row = idx / DD;
  int t = row % TT;
  float v = xn2[idx];
  if (t > 0) v = fmaxf(v, xn2[idx - DD]);
  if (t + 1 < TT) v = fmaxf(v, xn2[idx + DD]);
  mpx[idx] = v;
}

// pointwise conv D->32 from (B,T,D) layout; out[(b*outCh+chOff+f)*T + t]
__global__ void k_pw32(const float* __restrict__ in, const float* __restrict__ w,
                       float* __restrict__ out, int outCh, int chOff) {
  int idx = blockIdx.x * 256 + threadIdx.x;
  if (idx >= BB * NFNF * TT) return;
  int t = idx % TT;
  int f = (idx / TT) % NFNF;
  int b = idx / (TT * NFNF);
  const float* xr = in + ((size_t)b * TT + t) * DD;
  const float* wr = w + (size_t)f * DD;
  float acc = 0.0f;
  for (int d = 0; d < DD; ++d) acc += xr[d] * wr[d];
  out[((size_t)b * outCh + chOff + f) * TT + t] = acc;
}

// 32->32 conv1d, zero pad; writes into cat channel block chOff
__global__ void k_conv(const float* __restrict__ xb, const float* __restrict__ w,
                       float* __restrict__ cat, int Kw, int pad, int chOff) {
  int idx = blockIdx.x * 256 + threadIdx.x;
  if (idx >= BB * NFNF * TT) return;
  int t = idx % TT;
  int o = (idx / TT) % NFNF;
  int b = idx / (TT * NFNF);
  float acc = 0.0f;
  for (int i = 0; i < NFNF; ++i) {
    const float* xr = xb + ((size_t)b * NFNF + i) * TT;
    const float* wr = w + ((size_t)o * NFNF + i) * Kw;
    for (int k = 0; k < Kw; ++k) {
      int tt = t + k - pad;
      if (tt >= 0 && tt < TT) acc += xr[tt] * wr[k];
    }
  }
  cat[((size_t)b * 128 + chOff + o) * TT + t] = acc;
}

__global__ __launch_bounds__(256) void k_bnstats(const float* __restrict__ cat,
                                                 float* __restrict__ stats) {
  __shared__ float sh[256];
  int c = blockIdx.x, tid = threadIdx.x;
  float s = 0.0f, sq = 0.0f;
  for (int i = tid; i < BB * TT; i += 256) {
    int b = i / TT, t = i % TT;
    float v = cat[((size_t)b * 128 + c) * TT + t];
    s += v;
    sq += v * v;
  }
  float sum = blk_reduce256(s, sh, tid);
  float ssq = blk_reduce256(sq, sh, tid);
  if (tid == 0) {
    float mu = sum / (float)(BB * TT);
    float var = ssq / (float)(BB * TT) - mu * mu;
    stats[2 * c] = mu;
    stats[2 * c + 1] = rsqrtf(var + 1e-5f);
  }
}

__global__ void k_incep(const float* __restrict__ cat, const float* __restrict__ stats,
                        const float* __restrict__ g, const float* __restrict__ bb,
                        _Float16* __restrict__ inceph) {
  int idx = blockIdx.x * 256 + threadIdx.x;
  if (idx >= RR * 128) return;
  int c = idx & 127;
  int row = idx >> 7;
  int b = row / TT, t = row % TT;
  float v = cat[((size_t)b * 128 + c) * TT + t];
  float y = (v - stats[2 * c]) * stats[2 * c + 1] * g[c] + bb[c];
  inceph[(size_t)row * 128 + c] = (_Float16)gelu_f(y);
}

__global__ void k_loss(const float* __restrict__ gate_sum, float* __restrict__ out) {
  if (threadIdx.x == 0 && blockIdx.x == 0) {
    float s = 0.0f;
    for (int e = 0; e < EE; ++e) {
      float m = gate_sum[e] / (float)RR;
      s += m * m;
    }
    out[(size_t)RR * DD] = (float)EE * s;
  }
}

extern "C" void kernel_launch(void* const* d_in, const int* in_sizes, int n_in,
                              void* d_out, int out_size, void* d_ws, size_t ws_size,
                              hipStream_t stream) {
  (void)in_sizes; (void)n_in; (void)out_size; (void)ws_size;
  const float* x         = (const float*)d_in[0];
  // d_in[1] = left_k scalar (2048, compile-time constant here)
  const float* gamma1    = (const float*)d_in[2];
  const float* gamma2    = (const float*)d_in[3];
  const float* attn_w    = (const float*)d_in[4];
  const float* attn_b    = (const float*)d_in[5];
  const float* gate_w    = (const float*)d_in[6];
  const float* expert_w1 = (const float*)d_in[7];
  const float* expert_w2 = (const float*)d_in[8];
  const float* bott_w    = (const float*)d_in[9];
  const float* conv_w39  = (const float*)d_in[10];
  const float* conv_w19  = (const float*)d_in[11];
  const float* conv_w9   = (const float*)d_in[12];
  const float* mp_w      = (const float*)d_in[13];
  const float* bn_g      = (const float*)d_in[14];
  const float* bn_b      = (const float*)d_in[15];
  const float* proj_w    = (const float*)d_in[16];
  const float* proj_b    = (const float*)d_in[17];
  float* out = (float*)d_out;

  char* base = (char*)d_ws;
  size_t off = 0;
  auto alloc = [&](size_t bytes) -> char* {
    off = (off + 255) & ~(size_t)255;
    char* p = base + off;
    off += bytes;
    return p;
  };
  float*    xs       = (float*)alloc((size_t)BB * NN * DD * 4);
  float*    scores   = (float*)alloc((size_t)BB * NN * 4);
  int*      mask     = (int*)alloc((size_t)BB * NN * 4);
  int*      idxmap   = (int*)alloc((size_t)BB * KLEFT * 4);
  float*    extra    = (float*)alloc((size_t)BB * DD * 4);
  float*    xcat     = (float*)alloc((size_t)RR * DD * 4);
  float*    xn2      = (float*)alloc((size_t)RR * DD * 4);
  _Float16* xn2h     = (_Float16*)alloc((size_t)RR * DD * 2);
  float*    gate     = (float*)alloc((size_t)RR * EE * 4);
  float*    gate_sum = (float*)alloc(256);
  _Float16* w1h      = (_Float16*)alloc((size_t)EE * DD * HH * 2);
  _Float16* w2h      = (_Float16*)alloc((size_t)EE * HH * DD * 2);
  _Float16* projwh   = (_Float16*)alloc((size_t)128 * DD * 2);
  _Float16* hbuf     = (_Float16*)alloc((size_t)EE * RR * HH * 2);
  float*    moe_out  = (float*)alloc((size_t)RR * DD * 4);
  float*    xb       = (float*)alloc((size_t)BB * NFNF * TT * 4);
  float*    mpx      = (float*)alloc((size_t)RR * DD * 4);
  float*    cat      = (float*)alloc((size_t)BB * 128 * TT * 4);
  float*    stats    = (float*)alloc(256 * 4);
  _Float16* inceph   = (_Float16*)alloc((size_t)RR * 128 * 2);

  k_zero<<<1, 64, 0, stream>>>(gate_sum, 64);
  k_rms_score<<<BB * NN, 256, 0, stream>>>(x, gamma1, attn_w, attn_b, xs, scores);
  k_rank<<<BB * NN, 256, 0, stream>>>(scores, mask);
  k_compact<<<BB, 1024, 0, stream>>>(mask, idxmap);
  k_extra<<<BB, 512, 0, stream>>>(xs, mask, extra);
  k_xcat<<<(RR * DD + 255) / 256, 256, 0, stream>>>(xs, idxmap, extra, xcat);
  k_rms2<<<RR, 256, 0, stream>>>(xcat, gamma2, xn2);
  k_cvt<<<(RR * DD + 255) / 256, 256, 0, stream>>>(xn2, xn2h, RR * DD);
  k_cvt<<<(EE * DD * HH + 255) / 256, 256, 0, stream>>>(expert_w1, w1h, EE * DD * HH);
  k_cvt<<<(EE * HH * DD + 255) / 256, 256, 0, stream>>>(expert_w2, w2h, EE * HH * DD);
  k_cvt<<<(128 * DD + 255) / 256, 256, 0, stream>>>(proj_w, projwh, 128 * DD);
  k_gate<<<RR, 256, 0, stream>>>(xn2, gate_w, gate, gate_sum);
  k_gemm1<<<dim3(HH / BLK_N, (RR + BLK_M - 1) / BLK_M, EE), 256, 0, stream>>>(
      xn2h, w1h, hbuf, RR, DD, HH);
  k_gemm2<<<dim3(DD / BLK_N, (RR + BLK_M - 1) / BLK_M), 256, 0, stream>>>(
      hbuf, w2h, gate, moe_out, RR, HH, DD);
  k_mpx<<<(RR * DD + 255) / 256, 256, 0, stream>>>(xn2, mpx);
  k_pw32<<<(BB * NFNF * TT + 255) / 256, 256, 0, stream>>>(xn2, bott_w, xb, NFNF, 0);
  k_pw32<<<(BB * NFNF * TT + 255) / 256, 256, 0, stream>>>(mpx, mp_w, cat, 128, 96);
  k_conv<<<(BB * NFNF * TT + 255) / 256, 256, 0, stream>>>(xb, conv_w39, cat, 39, 19, 0);
  k_conv<<<(BB * NFNF * TT + 255) / 256, 256, 0, stream>>>(xb, conv_w19, cat, 19, 9, 32);
  k_conv<<<(BB * NFNF * TT + 255) / 256, 256, 0, stream>>>(xb, conv_w9, cat, 9, 4, 64);
  k_bnstats<<<128, 256, 0, stream>>>(cat, stats);
  k_incep<<<(RR * 128 + 255) / 256, 256, 0, stream>>>(cat, stats, bn_g, bn_b, inceph);
  k_gemmproj<<<dim3(DD / BLK_N, (RR + BLK_M - 1) / BLK_M), 256, 0, stream>>>(
      inceph, projwh, proj_b, xcat, moe_out, out, RR, 128, DD);
  k_loss<<<1, 1, 0, stream>>>(gate_sum, out);
}
